// LSTMModel_4002909520753
// MI455X (gfx1250) — matrix operand order repaired and checked by reading, not compiled, not run on hardware
//
#include <hip/hip_runtime.h>

typedef __attribute__((ext_vector_type(16))) _Float16 v16h;
typedef __attribute__((ext_vector_type(8)))  float    v8f;

#define B_TOT    16384
#define T_STEPS  60
#define H_DIM    50
#define G_DIM    200          // 4*H
#define NTILES   13           // ceil(200/16) -> covers gate cols 0..207
#define GSTRIDE  208          // padded gate row stride (floats)
#define WAVES    4
#define BLOCK    (WAVES * 32)
#define ROWS_PER_BLOCK (WAVES * 16)

// Augmented K layout (padded to 64):
//   K = 0..49  : recurrent h
//   K = 50     : x_t            (B row = w_ih[g])
//   K = 51     : constant 1.0   (B row = b_ih[g] + b_hh[g])
//   K = 52..63 : zero pad
// => WMMA directly produces the full pre-activation gates.

// ---- dynamic LDS layout (bytes) ----
// wfrag : 13 tiles * 2 kchunks * 32 lanes * 16 halfs * 2B = 26624
// gates : WAVES * 16 * 208 * 4                            = 53248
// hfrag : WAVES * 2 * 32 * 16 * 2                         =  8192
// wlin  : 3000 * 4                                        = 12000
#define OFF_WFRAG 0
#define OFF_GATES 26624
#define OFF_HFRAG (OFF_GATES + WAVES * 16 * GSTRIDE * 4)       // 79872
#define OFF_WLIN  (OFF_HFRAG + WAVES * 2 * 32 * 16 * 2)        // 88064
#define SMEM_BYTES (OFF_WLIN + 3000 * 4)                       // 100064

#if __has_builtin(__builtin_amdgcn_tanhf)
__device__ __forceinline__ float t_tanh(float x) { return __builtin_amdgcn_tanhf(x); }
#else
__device__ __forceinline__ float t_tanh(float x) { return 1.0f - 2.0f / (__expf(2.0f * x) + 1.0f); }
#endif
__device__ __forceinline__ float t_sig(float x) { return 0.5f * t_tanh(0.5f * x) + 0.5f; }

__global__ __launch_bounds__(BLOCK) void lstm_fused_kernel(
    const float* __restrict__ input,   // [B, T] (IN==1)
    const float* __restrict__ w_ih,    // [200]  (IN==1)
    const float* __restrict__ w_hh,    // [200, 50]
    const float* __restrict__ b_ih,    // [200]
    const float* __restrict__ b_hh,    // [200]
    const float* __restrict__ w_lin,   // [3000] = [T*H]
    const float* __restrict__ b_lin,   // [1]
    float* __restrict__ out)           // [B]
{
    extern __shared__ char smem[];
    _Float16* wfrag = (_Float16*)(smem + OFF_WFRAG);
    float*    gates = (float*)   (smem + OFF_GATES);
    _Float16* hfrag = (_Float16*)(smem + OFF_HFRAG);
    float*    wlinS = (float*)   (smem + OFF_WLIN);

    const int tid = threadIdx.x;

    // ---- init: pack augmented B matrix [w_hh | w_ih | bias | 0] into fragment layout ----
    // B[k][n] for tile nt, chunk ck: lane = n + 16*((k/8)&1); half index = (k/16)*8 + (k%8)
    for (int idx = tid; idx < NTILES * 2 * 32 * 16; idx += BLOCK) {
        int tile = idx / 1024;
        int rem  = idx - tile * 1024;
        int ck   = rem >> 9;          // 0..1
        int r2   = rem & 511;
        int k    = r2 >> 4;           // 0..31
        int n    = r2 & 15;
        int j    = ck * 32 + k;       // augmented K index
        int g    = tile * 16 + n;
        float v = 0.0f;
        if (g < G_DIM) {
            if (j < H_DIM)        v = w_hh[g * H_DIM + j];
            else if (j == 50)     v = w_ih[g];
            else if (j == 51)     v = b_ih[g] + b_hh[g];
        }
        int lane = n + 16 * ((k >> 3) & 1);
        int half = ((k >> 4) << 3) + (k & 7);
        wfrag[((tile * 2 + ck) * 32 + lane) * 16 + half] = (_Float16)v;
    }
    for (int idx = tid; idx < T_STEPS * H_DIM; idx += BLOCK) wlinS[idx] = w_lin[idx];
    // zero h fragments (h0 = 0; pad K stays zero forever)
    for (int idx = tid; idx < WAVES * 2 * 32 * 16; idx += BLOCK)
        hfrag[idx] = (_Float16)0.0f;
    __syncthreads();
    // set constant-1.0 at augmented K=51 for every row of every wave's A fragment:
    // k=51 -> ck=1, kl=19 -> lane = row, half = 8 + 3 = 11
    for (int idx = tid; idx < WAVES * 16; idx += BLOCK) {
        int w = idx >> 4, m = idx & 15;
        hfrag[w * 2 * 32 * 16 + (1 * 32 + m) * 16 + 11] = (_Float16)1.0f;
    }
    __syncthreads();

    const int wave = tid >> 5;
    const int lane = tid & 31;
    const int bl   = lane & 15;   // local batch row (and N-col inside a tile)
    const int hi   = lane >> 4;   // lane half-group
    const int bbase = blockIdx.x * ROWS_PER_BLOCK + wave * 16;

    float*    gw = gates + wave * 16 * GSTRIDE;
    _Float16* hw = hfrag + wave * 2 * 32 * 16;

    float creg[25];
#pragma unroll
    for (int i = 0; i < 25; ++i) creg[i] = 0.0f;
    float oacc = 0.0f;

    for (int t = 0; t < T_STEPS; ++t) {
        // ---- inject x_t into augmented A at K=50 (ck=1, kl=18 -> lane=row, half=10) ----
        const float x = input[(bbase + bl) * T_STEPS + t];
        if (hi == 0)
            hw[(1 * 32 + bl) * 16 + 10] = (_Float16)x;

        // ---- load [h | x | 1] as two A fragments (M=16 batch, augK 0..31 / 32..63) ----
        v16h a0 = *(const v16h*)(hw + (0 * 32 + lane) * 16);
        v16h a1 = *(const v16h*)(hw + (1 * 32 + lane) * 16);

        // ---- full pre-activation gates via WMMA: 13 N-tiles x 2 chained K-chunks ----
        for (int nt = 0; nt < NTILES; ++nt) {
            v16h b0 = *(const v16h*)(wfrag + ((nt * 2 + 0) * 32 + lane) * 16);
            v16h b1 = *(const v16h*)(wfrag + ((nt * 2 + 1) * 32 + lane) * 16);
            v8f acc = {};
            acc = __builtin_amdgcn_wmma_f32_16x16x32_f16(
                false, a0, false, b0, (short)0, acc, false, false);
            acc = __builtin_amdgcn_wmma_f32_16x16x32_f16(
                false, a1, false, b1, (short)0, acc, false, false);
            // D layout: VGPR r -> M = r + 8*hi, N = bl
            int gcol = nt * 16 + bl;
#pragma unroll
            for (int r = 0; r < 8; ++r)
                gw[(r + 8 * hi) * GSTRIDE + gcol] = acc[r];
        }

        // ---- activations + cell update + fused output accumulation ----
        const float* wl = wlinS + t * H_DIM;
#pragma unroll
        for (int it = 0; it < 25; ++it) {
            const int j = 2 * it + hi;       // hidden index 0..49
            float gi = gw[bl * GSTRIDE + j];
            float gf = gw[bl * GSTRIDE + 50 + j];
            float gg = gw[bl * GSTRIDE + 100 + j];
            float go = gw[bl * GSTRIDE + 150 + j];
            float ig = t_sig(gi);
            float fg = t_sig(gf);
            float og = t_sig(go);
            float gt = t_tanh(gg);
            float c  = fg * creg[it] + ig * gt;
            creg[it] = c;
            float h  = og * t_tanh(c);
            // store h into A-fragment layout for next step:
            // A[m][k]: lane = m + 16*((k/8)&1); half = (k/16)*8 + (k%8)
            int kl = j & 31;
            int ck = j >> 5;
            int lt = bl + 16 * ((kl >> 3) & 1);
            int hf = ((kl >> 4) << 3) + (kl & 7);
            hw[(ck * 32 + lt) * 16 + hf] = (_Float16)h;
            // fused final linear: out[b] += h * w_lin[t*H + j]
            oacc += h * wl[j];
        }
        // all LDS traffic is intra-wave (DS ops are in-order per wave): no barrier
    }

    // ---- reduce partial outputs: lane l and l+16 hold the two j-parities of row bl ----
    gw[lane] = oacc;
    __syncthreads();
    if (lane < 16)
        out[bbase + lane] = gw[lane] + gw[lane + 16] + b_lin[0];
}

extern "C" void kernel_launch(void* const* d_in, const int* in_sizes, int n_in,
                              void* d_out, int out_size, void* d_ws, size_t ws_size,
                              hipStream_t stream) {
    (void)in_sizes; (void)n_in; (void)d_ws; (void)ws_size; (void)out_size;
    const float* input = (const float*)d_in[0];   // [16384, 60, 1]
    const float* w_ih  = (const float*)d_in[1];   // [200, 1]
    const float* w_hh  = (const float*)d_in[2];   // [200, 50]
    const float* b_ih  = (const float*)d_in[3];   // [200]
    const float* b_hh  = (const float*)d_in[4];   // [200]
    const float* w_lin = (const float*)d_in[5];   // [1, 3000]
    const float* b_lin = (const float*)d_in[6];   // [1]
    float* out = (float*)d_out;                   // [16384, 1]

    dim3 grid(B_TOT / ROWS_PER_BLOCK);            // 256 blocks
    dim3 block(BLOCK);                            // 128 threads = 4 waves
    lstm_fused_kernel<<<grid, block, SMEM_BYTES, stream>>>(
        input, w_ih, w_hh, b_ih, b_hh, w_lin, b_lin, out);
}
